// CKDLoss_8014408974769
// MI455X (gfx1250) — matrix operand order repaired and checked by reading, not compiled, not run on hardware
//
#include <hip/hip_runtime.h>
#include <hip/hip_bf16.h>

typedef __attribute__((ext_vector_type(16))) __bf16 v16bf;
typedef __attribute__((ext_vector_type(8)))  float  v8f;

#define B_ROWS   64
#define C_COLS   100
#define K_TEMPS  5
#define N_FLAT   (B_ROWS * C_COLS * K_TEMPS)   // 32000
#define CPAD     128                           // C padded to 4 K-chunks of 32
#define MPAD     112                           // 100 padded to 7 tiles of 16
#define ALPHA_C  0.7f

// ---------------- workspace layout (float elements) ----------------
// f[0..7]  : accumulators
//   0: kl weighted sum   1: ce sum   2: tt   3: ss   4: ts   5: l1   6: gram
// f[8 ..)                : tc32 [K][B][C]   (32000)
// next                   : sc32 [K][B][C]   (32000)
// then bf16 region:
//   Tb   [K][64][128]  teacher softmax (padded)
//   Sb   [K][64][128]  student softmax (padded)
//   Sng  [K][64][128]  -student
//   Tt   [K][112][64]  teacher transposed
//   Sngt [K][112][64]  -student transposed
#define BF_TB    0
#define BF_SB    (K_TEMPS * B_ROWS * CPAD)
#define BF_SNG   (2 * K_TEMPS * B_ROWS * CPAD)
#define BF_TT    (3 * K_TEMPS * B_ROWS * CPAD)
#define BF_SNGT  (3 * K_TEMPS * B_ROWS * CPAD + K_TEMPS * MPAD * B_ROWS)
#define BF_TOTAL (3 * K_TEMPS * B_ROWS * CPAD + 2 * K_TEMPS * MPAD * B_ROWS) // 194560
#define BF_WORDS (BF_TOTAL / 2)                                              // 97280

// ---------------------------------------------------------------- zero
__global__ void ckd_zero_kernel(float* accum, unsigned int* bz, int nwords) {
    int i = blockIdx.x * blockDim.x + threadIdx.x;
    if (i < 8) accum[i] = 0.0f;
    for (int j = i; j < nwords; j += gridDim.x * blockDim.x) bz[j] = 0u;
}

// ------------------------------------------------- block reduction helpers
__device__ __forceinline__ float blk_max128(float* red, float v) {
    int tid = threadIdx.x;
    __syncthreads();
    red[tid] = v; __syncthreads();
    for (int s = 64; s > 0; s >>= 1) {
        if (tid < s) red[tid] = fmaxf(red[tid], red[tid + s]);
        __syncthreads();
    }
    float r = red[0];
    return r;
}
__device__ __forceinline__ float blk_sum128(float* red, float v) {
    int tid = threadIdx.x;
    __syncthreads();
    red[tid] = v; __syncthreads();
    for (int s = 64; s > 0; s >>= 1) {
        if (tid < s) red[tid] = red[tid] + red[tid + s];
        __syncthreads();
    }
    float r = red[0];
    return r;
}

// -------------------------------------------------- softmax / KD / cubes
// grid = (B_ROWS, K_TEMPS), block = 128. One block per (batch row, temp).
__global__ void ckd_softmax_kernel(const float* __restrict__ ls,
                                   const float* __restrict__ lt,
                                   const int*   __restrict__ target,
                                   float* tc32, float* sc32,
                                   __bf16* bw, float* accum) {
    __shared__ float red[128];
    const int b = blockIdx.x;
    const int k = blockIdx.y;            // temp = k + 1
    const float temp = (float)(k + 1);
    const int tid = threadIdx.x;
    const bool on = (tid < C_COLS);

    float xs = on ? ls[b * C_COLS + tid] / temp : -1e30f;
    float xt = on ? lt[b * C_COLS + tid] / temp : -1e30f;

    float maxs = blk_max128(red, xs);
    float maxt = blk_max128(red, xt);
    float es = on ? expf(xs - maxs) : 0.0f;
    float et = on ? expf(xt - maxt) : 0.0f;
    float sums = blk_sum128(red, es);
    float sumt = blk_sum128(red, et);

    float log_q = xs - maxs - logf(sums);   // student log-softmax
    float log_p = xt - maxt - logf(sumt);   // teacher log-softmax
    float q = es / sums;                    // student softmax
    float p = et / sumt;                    // teacher softmax

    float klpart = 0.0f;
    if (on) {
        int idx = (k * B_ROWS + b) * C_COLS + tid;
        tc32[idx] = p;
        sc32[idx] = q;
        // bf16 copies for WMMA Gram kernels
        int r = (k * B_ROWS + b) * CPAD + tid;       // [K][64][128]
        bw[BF_TB  + r] = (__bf16)p;
        bw[BF_SB  + r] = (__bf16)q;
        bw[BF_SNG + r] = (__bf16)(-q);
        int rt = (k * MPAD + tid) * B_ROWS + b;      // [K][112][64]
        bw[BF_TT   + rt] = (__bf16)p;
        bw[BF_SNGT + rt] = (__bf16)(-q);
        klpart = p * (log_p - log_q);
    }
    float klsum = blk_sum128(red, klpart);
    if (tid == 0)
        atomicAdd(&accum[0], klsum * (ALPHA_C * temp * temp) /
                              (float)(B_ROWS * C_COLS));
    // CE uses temp==1 student log-softmax
    if (k == 0 && on && tid == target[b])
        atomicAdd(&accum[1], -log_q);
}

// ---------------------------------------------------- tt / ss / ts dots
__global__ void ckd_dots_kernel(const float* __restrict__ t,
                                const float* __restrict__ s, float* accum) {
    __shared__ float red[256];
    float att = 0.f, ass = 0.f, ats = 0.f;
    for (int i = blockIdx.x * blockDim.x + threadIdx.x; i < N_FLAT;
         i += gridDim.x * blockDim.x) {
        float tv = t[i], sv = s[i];
        att += tv * tv; ass += sv * sv; ats += tv * sv;
    }
    for (int w = 0; w < 3; ++w) {
        float v = (w == 0) ? att : (w == 1) ? ass : ats;
        __syncthreads();
        red[threadIdx.x] = v; __syncthreads();
        for (int sft = 128; sft > 0; sft >>= 1) {
            if (threadIdx.x < sft) red[threadIdx.x] += red[threadIdx.x + sft];
            __syncthreads();
        }
        if (threadIdx.x == 0) atomicAdd(&accum[2 + w], red[0]);
    }
}

// --------------------------------------- L1 over virtual N x N outer product
// 125 blocks x 256 threads; each thread owns one 'a' index; b streams via LDS.
#define L1_CHUNK 1280
__global__ void ckd_l1_kernel(const float* __restrict__ t,
                              const float* __restrict__ s, float* accum) {
    __shared__ float tl[L1_CHUNK];
    __shared__ float sl[L1_CHUNK];
    __shared__ float red[256];
    const int a = blockIdx.x * 256 + threadIdx.x;    // 125*256 == 32000
    const float ta = t[a], sa = s[a];
    float acc = 0.0f;
    for (int base = 0; base < N_FLAT; base += L1_CHUNK) {
        for (int j = threadIdx.x; j < L1_CHUNK; j += 256) {
            tl[j] = t[base + j];
            sl[j] = s[base + j];
        }
        __syncthreads();
#pragma unroll 8
        for (int j = 0; j < L1_CHUNK; ++j)
            acc += fabsf(fmaf(ta, tl[j], -(sa * sl[j])));
        __syncthreads();
    }
    red[threadIdx.x] = acc; __syncthreads();
    for (int sft = 128; sft > 0; sft >>= 1) {
        if (threadIdx.x < sft) red[threadIdx.x] += red[threadIdx.x + sft];
        __syncthreads();
    }
    if (threadIdx.x == 0) atomicAdd(&accum[5], red[0]);
}

// ------------------------------------------------------- WMMA fragment loads
// A fragment: 16x32 bf16, rows m0.., row-major source with stride lda.
__device__ __forceinline__ v16bf load_a_frag(const __bf16* src, int m0,
                                             int k0, int lda) {
    const int l = threadIdx.x;
    const int m = m0 + (l & 15);
    const int half = l >> 4;
    v16bf a;
#pragma unroll
    for (int v = 0; v < 8; ++v) {
        int kk = k0 + half * 16 + 2 * v;
        a[2 * v]     = src[m * lda + kk];
        a[2 * v + 1] = src[m * lda + kk + 1];
    }
    return a;
}
// B fragment: 32x16 bf16 taken as the TRANSPOSE of a row-major source:
// B[kk][n] = src[n*lda + kk]  (i.e. B = src^T restricted to cols n0..)
__device__ __forceinline__ v16bf load_bT_frag(const __bf16* src, int n0,
                                              int k0, int lda) {
    const int l = threadIdx.x;
    const int n = n0 + (l & 15);
    const int half = l >> 4;
    v16bf b;
#pragma unroll
    for (int v = 0; v < 8; ++v) {
        int kk = k0 + half * 16 + 2 * v;
        b[2 * v]     = src[n * lda + kk];
        b[2 * v + 1] = src[n * lda + kk + 1];
    }
    return b;
}
// B fragment from a row-major [K x N] source directly: B[kk][n] = src[kk*ldb+n]
__device__ __forceinline__ v16bf load_b_frag(const __bf16* src, int n0,
                                             int k0, int ldb) {
    const int l = threadIdx.x;
    const int n = n0 + (l & 15);
    const int half = l >> 4;
    v16bf b;
#pragma unroll
    for (int v = 0; v < 8; ++v) {
        int kk = k0 + half * 16 + 2 * v;
        b[2 * v]     = src[kk * ldb + n];
        b[2 * v + 1] = src[(kk + 1) * ldb + n];
    }
    return b;
}

__device__ __forceinline__ void frob_reduce_add(v8f acc, float* dst) {
    __shared__ float red[32];
    float ssum = 0.0f;
#pragma unroll
    for (int i = 0; i < 8; ++i) ssum += acc[i] * acc[i];
    red[threadIdx.x] = ssum;
    __syncthreads();
    if (threadIdx.x == 0) {
        float tot = 0.0f;
#pragma unroll
        for (int i = 0; i < 32; ++i) tot += red[i];
        atomicAdd(dst, tot);
    }
}

// ---- G_k = T_k T_k^T - S_k S_k^T  (64x64, K=100 padded to 128) ----
// grid = K_TEMPS * 16 waves, 32 threads each; one 16x16 tile per wave.
__global__ void ckd_gram_g_kernel(const __bf16* __restrict__ bw, float* accum) {
    const int k  = blockIdx.x >> 4;
    const int t  = blockIdx.x & 15;
    const int m0 = (t >> 2) * 16;
    const int n0 = (t & 3) * 16;
    const __bf16* T  = bw + BF_TB  + k * B_ROWS * CPAD;
    const __bf16* S  = bw + BF_SB  + k * B_ROWS * CPAD;
    const __bf16* Sn = bw + BF_SNG + k * B_ROWS * CPAD;
    v8f acc = {};
#pragma unroll
    for (int kc = 0; kc < CPAD; kc += 32) {        // + T T^T
        v16bf a = load_a_frag(T, m0, kc, CPAD);
        v16bf b = load_bT_frag(T, n0, kc, CPAD);
        acc = __builtin_amdgcn_wmma_f32_16x16x32_bf16(false, a, false, b,
                                                      (short)0, acc, false, false);
    }
#pragma unroll
    for (int kc = 0; kc < CPAD; kc += 32) {        // + (-S) S^T
        v16bf a = load_a_frag(Sn, m0, kc, CPAD);
        v16bf b = load_bT_frag(S, n0, kc, CPAD);
        acc = __builtin_amdgcn_wmma_f32_16x16x32_bf16(false, a, false, b,
                                                      (short)0, acc, false, false);
    }
    frob_reduce_add(acc, &accum[6]);
}

// ---- H_k = T_k^T T_k - S_k^T S_k  (100x100 padded to 112, K=64) ----
// grid = K_TEMPS * 49 waves; A from transposed arrays [112][64], B from
// the row-major [64][128] arrays (n < 112 <= 128, zero-padded).
__global__ void ckd_gram_h_kernel(const __bf16* __restrict__ bw, float* accum) {
    const int k  = blockIdx.x / 49;
    const int t  = blockIdx.x % 49;
    const int m0 = (t / 7) * 16;
    const int n0 = (t % 7) * 16;
    const __bf16* Tt  = bw + BF_TT   + k * MPAD * B_ROWS;
    const __bf16* Snt = bw + BF_SNGT + k * MPAD * B_ROWS;
    const __bf16* T   = bw + BF_TB   + k * B_ROWS * CPAD;
    const __bf16* S   = bw + BF_SB   + k * B_ROWS * CPAD;
    v8f acc = {};
#pragma unroll
    for (int kc = 0; kc < B_ROWS; kc += 32) {      // + T^T T
        v16bf a = load_a_frag(Tt, m0, kc, B_ROWS);
        v16bf b = load_b_frag(T, n0, kc, CPAD);
        acc = __builtin_amdgcn_wmma_f32_16x16x32_bf16(false, a, false, b,
                                                      (short)0, acc, false, false);
    }
#pragma unroll
    for (int kc = 0; kc < B_ROWS; kc += 32) {      // + (-S^T) S
        v16bf a = load_a_frag(Snt, m0, kc, B_ROWS);
        v16bf b = load_b_frag(S, n0, kc, CPAD);
        acc = __builtin_amdgcn_wmma_f32_16x16x32_bf16(false, a, false, b,
                                                      (short)0, acc, false, false);
    }
    frob_reduce_add(acc, &accum[6]);
}

// ---------------------------------------------------------------- finalize
__global__ void ckd_finalize_kernel(const float* accum, float* out) {
    float ce = accum[1] / (float)B_ROWS;
    float kd = accum[0] + (float)K_TEMPS * (1.0f - ALPHA_C) * ce;
    float tt = accum[2], ss = accum[3], ts = accum[4];
    float l2 = 0.00025f * (tt * tt - 2.0f * ts * ts + ss * ss);
    float l1 = 0.00025f * accum[5];
    out[0] = kd + l1 + l2 + accum[6];
}

// ------------------------------------------------------------------ launch
extern "C" void kernel_launch(void* const* d_in, const int* in_sizes, int n_in,
                              void* d_out, int out_size, void* d_ws, size_t ws_size,
                              hipStream_t stream) {
    const float* ls     = (const float*)d_in[0];
    const float* lt     = (const float*)d_in[1];
    const int*   target = (const int*)d_in[2];
    float* out = (float*)d_out;

    float* f     = (float*)d_ws;
    float* accum = f;
    float* tc32  = f + 8;
    float* sc32  = tc32 + N_FLAT;
    __bf16* bw   = (__bf16*)(sc32 + N_FLAT);

    ckd_zero_kernel<<<128, 256, 0, stream>>>(accum, (unsigned int*)bw, BF_WORDS);

    dim3 sgrid(B_ROWS, K_TEMPS);
    ckd_softmax_kernel<<<sgrid, 128, 0, stream>>>(ls, lt, target, tc32, sc32,
                                                  bw, accum);

    ckd_dots_kernel<<<64, 256, 0, stream>>>(tc32, sc32, accum);

    ckd_l1_kernel<<<N_FLAT / 256, 256, 0, stream>>>(tc32, sc32, accum);

    ckd_gram_g_kernel<<<K_TEMPS * 16, 32, 0, stream>>>(bw, accum);
    ckd_gram_h_kernel<<<K_TEMPS * 49, 32, 0, stream>>>(bw, accum);

    ckd_finalize_kernel<<<1, 1, 0, stream>>>(accum, out);
}
